// ContinualSVGP_62328565399553
// MI455X (gfx1250) — compile-verified, matrix operand order checked by reading, not run on hardware
//
#include <hip/hip_runtime.h>
#include <hip/hip_bf16.h>

// ---------------------------------------------------------------------------
// Continual SVGP predictive for MI455X (gfx1250, wave32, WMMA).
// H=4, O=32, M=128, D=16, B=1024.  All GEMM-shaped einsums (RBF cross terms,
// covariance assembly, vec2) run on v_wmma_f32_16x16x32_f16 (f16 in, f32 acc)
// with 64x64 block tiles, b128 LDS staging and b128 fragment loads.
// Feature dim padded 16->32 so every K is a whole 32-chunk: staging loads are
// fully unconditional (no exec masking) with running-pointer addressing.
// Cholesky / forward substitution stay in f32 VALU (serial recurrences).
// ---------------------------------------------------------------------------

constexpr int CH = 4, CO = 32, CM = 128, CD = 16, CB = 1024;
constexpr int CM2 = 2 * CM;          // 256
constexpr int CDP = 32;              // padded feature dim (zero tail)
constexpr float JIT = 1e-4f;

typedef __attribute__((ext_vector_type(16))) _Float16 v16h;
typedef __attribute__((ext_vector_type(8)))  _Float16 h8;
typedef __attribute__((ext_vector_type(8)))  float    v8f;

// -------- workspace layout (float elements) --------------------------------
constexpr long WS_SF2   = 0;                                   //      4
constexpr long WS_LSINV = 16;                                  //     64
constexpr long WS_ZS    = 128;                                 // H*O*2M*DP = 1048576
constexpr long WS_XSC   = WS_ZS  + (long)CH*CO*CM2*CDP;        // H*B*DP  = 131072
constexpr long WS_NZ    = WS_XSC + (long)CH*CB*CDP;            // H*O*2M  = 32768
constexpr long WS_NX    = WS_NZ  + (long)CH*CO*CM2;            // H*B     = 4096
constexpr long WS_MJ    = WS_NX  + (long)CH*CB;                // H*O*2M  = 32768
constexpr long WS_LKM   = WS_MJ  + (long)CH*CO*CM2;            // H*O*M   = 16384
constexpr long WS_AM    = WS_LKM + (long)CH*CO*CM;             // H*O*M   = 16384
constexpr long WS_COV   = 2097152;                             // H*O*2M*2M = 8388608
constexpr long WS_KUU2  = WS_COV  + (long)CH*CO*CM2*CM2;       // 8388608
constexpr long WS_KUU   = WS_KUU2 + (long)CH*CO*CM2*CM2;       // 2097152
constexpr long WS_KUF   = WS_KUU  + (long)CH*CO*CM*CM;         // 2097152
constexpr long WS_LLS   = WS_KUF  + (long)CH*CO*CM*CM;         // 2097152
constexpr long WS_ATX   = WS_LLS  + (long)CH*CO*CM*CM;         // 2097152
constexpr long WS_UTRIL = WS_ATX  + (long)CH*CO*CM*CM;         // O*M*M = 524288
// kuf2 (H*O*2M*B = 33554432) overlays KUU..UTRIL, which are dead by then.
constexpr long WS_KUF2  = WS_KUU;

// ---------------------------------------------------------------------------
// prep kernels
// ---------------------------------------------------------------------------
__global__ void prep_theta_kernel(const float* __restrict__ theta,
                                  float* __restrict__ sf2,
                                  float* __restrict__ lsinv) {
  int t = threadIdx.x;
  if (t < CH) sf2[t] = expf(theta[t * (CD + 1)]);
  if (t < CH * CD) {
    int h = t / CD, d = t % CD;
    lsinv[t] = expf(-theta[h * (CD + 1) + 1 + d]);
  }
}

// scaled z_joint rows [h][o][2M][DP] (zero-padded) + squared norms
__global__ void prep_zs_kernel(const float* __restrict__ z,
                               const float* __restrict__ z_old,
                               const float* __restrict__ lsinv,
                               float* __restrict__ Zs, float* __restrict__ nZ) {
  int r = blockIdx.x * blockDim.x + threadIdx.x;
  if (r >= CH * CO * CM2) return;
  int h = r / (CO * CM2);
  int rem = r % (CO * CM2);
  int o = rem / CM2;
  int idx = rem % CM2;
  const float* src = (idx < CM) ? (z_old + ((long)o * CM + idx) * CD)
                                : (z     + ((long)o * CM + (idx - CM)) * CD);
  float nrm = 0.f;
  for (int d = 0; d < CD; ++d) {
    float v = src[d] * lsinv[h * CD + d];
    Zs[(long)r * CDP + d] = v;
    nrm += v * v;
  }
  for (int d = CD; d < CDP; ++d) Zs[(long)r * CDP + d] = 0.f;
  nZ[r] = nrm;
}

__global__ void prep_x_kernel(const float* __restrict__ x,
                              const float* __restrict__ lsinv,
                              float* __restrict__ Xs, float* __restrict__ nX) {
  int r = blockIdx.x * blockDim.x + threadIdx.x;
  if (r >= CH * CB) return;
  int h = r / CB, n = r % CB;
  float nrm = 0.f;
  for (int d = 0; d < CD; ++d) {
    float v = x[(long)n * CD + d] * lsinv[h * CD + d];
    Xs[(long)r * CDP + d] = v;
    nrm += v * v;
  }
  for (int d = CD; d < CDP; ++d) Xs[(long)r * CDP + d] = 0.f;
  nX[r] = nrm;
}

__global__ void prep_utril_kernel(const float* __restrict__ vec,
                                  float* __restrict__ U) {
  int t = blockIdx.x * blockDim.x + threadIdx.x;   // o*M + row
  if (t >= CO * CM) return;
  int o = t / CM, r = t % CM;
  const float* vrow = vec + (long)o * ((long)CM * (CM + 1) / 2) + (long)r * (r + 1) / 2;
  float* urow = U + (long)o * CM * CM + (long)r * CM;
  for (int c = 0; c < CM; ++c) urow[c] = (c <= r) ? vrow[c] : 0.f;
}

// dst[(h*O+o)*dstStride + i] = src[o*cnt + i], i < cnt
__global__ void bcast_copy_kernel(const float* __restrict__ src,
                                  float* __restrict__ dst,
                                  long cnt, long dstStride, long total) {
  long t = (long)blockIdx.x * blockDim.x + threadIdx.x;
  long stride = (long)gridDim.x * blockDim.x;
  for (; t < total; t += stride) {
    long ho = t / cnt, i = t % cnt;
    long o = ho % CO;
    dst[ho * dstStride + i] = src[o * cnt + i];
  }
}

__global__ void add_mnew_kernel(const float* __restrict__ Am,
                                const float* __restrict__ u_mean,
                                float* __restrict__ mj) {
  int t = blockIdx.x * blockDim.x + threadIdx.x;
  if (t >= CH * CO * CM) return;
  int ho = t / CM, i = t % CM;
  int o = ho % CO;
  mj[(long)ho * CM2 + CM + i] = Am[(long)ho * CM + i] + u_mean[(long)o * CM + i];
}

// ---------------------------------------------------------------------------
// Fragment helper: lane fragment from an [row][k] LDS tile (row stride 40
// halves = 80B, so each 8-half group is 16B aligned -> ds_load_b128).
// v16h element e (e<8): K = 8*hi + e ; (e>=8): K = 16 + 8*hi + (e-8).
// ---------------------------------------------------------------------------
__device__ __forceinline__ v16h frag_from_lds(const _Float16* rowPtr, int hi) {
  h8 lo = *(const h8*)(rowPtr + hi * 8);
  h8 hiv = *(const h8*)(rowPtr + 16 + hi * 8);
  return __builtin_shufflevector(lo, hiv, 0, 1, 2, 3, 4, 5, 6, 7,
                                 8, 9, 10, 11, 12, 13, 14, 15);
}

// ---------------------------------------------------------------------------
// Generic batched strided WMMA GEMM:  S[m,n] = sum_k A[k,m]*B[k,n]  (f16 mul,
// f32 acc), 64x64 block tile, 4 waves x (2x2) 16x16 WMMA sub-tiles.
// REQUIREMENTS (guaranteed by caller): Kdim % 32 == 0; all staged A rows
// (m0..m0+63) and B element addresses are readable (Mrows % 64 == 0; for
// Ncols < 64 the caller must use sBn == 0 so column reads alias column 0 —
// out-of-range columns are computed but never stored).
// mode 0: C = S (+jitter on diag) (+= old C if accumulate).
// mode 1 (RBF): C = sf2[h]*exp(-0.5*max(nA[m]+nB[n]-2S, 0)) + jitter*(m==n)
// ---------------------------------------------------------------------------
__global__ __launch_bounds__(128)
void wmma_gemm_kernel(const float* __restrict__ Aq, const float* __restrict__ Bq,
                      float* __restrict__ Cq,
                      int Mrows, int Ncols, int Kdim,
                      long aSh, long aSo, long sAk, long sAm,
                      long bSh, long bSo, long sBk, long sBn,
                      long cSh, long cSo, long ldC,
                      int accumulate, float jitter, int mode,
                      const float* __restrict__ nAq, long naSh, long naSo,
                      const float* __restrict__ nBq, long nbSh, long nbSo,
                      const float* __restrict__ sf2v) {
  __shared__ alignas(16) _Float16 lA[64][40];   // [m][k]
  __shared__ alignas(16) _Float16 lB[64][40];   // [n][k]
  const int tid = threadIdx.x;
  const int h = blockIdx.z / CO, o = blockIdx.z % CO;
  const float* Ab = Aq + h * aSh + o * aSo;
  const float* Bb = Bq + h * bSh + o * bSo;
  float* Cb = Cq + h * cSh + o * cSo;
  const int m0 = blockIdx.x * 64, n0 = blockIdx.y * 64;
  const int lane = tid & 31, w = tid >> 5;
  const int wm = (w >> 1) * 32, wn = (w & 1) * 32;
  const int l16 = lane & 15, hi = lane >> 4;
  // staging assignment: each thread owns one row (0..63) and 16 k's
  const int srow = tid >> 1;
  const int skg  = (tid & 1) * 16;

  v8f acc00 = {0.f,0.f,0.f,0.f,0.f,0.f,0.f,0.f};
  v8f acc01 = acc00, acc10 = acc00, acc11 = acc00;

  // running pointers over k (stride sAk/sBk), advanced 32 per chunk
  const float* pA = Ab + (long)skg * sAk + (long)(m0 + srow) * sAm;
  const float* pB = Bb + (long)skg * sBk + (long)(n0 + srow) * sBn;
  const long stepA = 32 * sAk - 16 * sAk;   // after 16 incremental adds
  const long stepB = 32 * sBk - 16 * sBk;

  for (int k0 = 0; k0 < Kdim; k0 += 32) {
    // ---- stage 64x32 A and B tiles: fully unconditional clause loads ----
    {
      float fa[16], fb[16];
      const float* qa = pA;
      const float* qb = pB;
#pragma unroll
      for (int e = 0; e < 16; ++e) {
        fa[e] = *qa;  qa += sAk;
        fb[e] = *qb;  qb += sBk;
      }
      pA += 32 * sAk;
      pB += 32 * sBk;
      h8 ha0, ha1, hb0, hb1;
#pragma unroll
      for (int e = 0; e < 8; ++e) {
        ha0[e] = (_Float16)fa[e];     ha1[e] = (_Float16)fa[8 + e];
        hb0[e] = (_Float16)fb[e];     hb1[e] = (_Float16)fb[8 + e];
      }
      *(h8*)&lA[srow][skg]     = ha0;
      *(h8*)&lA[srow][skg + 8] = ha1;
      *(h8*)&lB[srow][skg]     = hb0;
      *(h8*)&lB[srow][skg + 8] = hb1;
    }
    __syncthreads();
    // ---- fragments: 2 A rows x 2 B rows -> 4 WMMAs ----
    v16h af0 = frag_from_lds(&lA[wm + l16][0],      hi);
    v16h af1 = frag_from_lds(&lA[wm + 16 + l16][0], hi);
    v16h bf0 = frag_from_lds(&lB[wn + l16][0],      hi);
    v16h bf1 = frag_from_lds(&lB[wn + 16 + l16][0], hi);
    acc00 = __builtin_amdgcn_wmma_f32_16x16x32_f16(false, af0, false, bf0,
                                                   (short)0, acc00, false, false);
    acc01 = __builtin_amdgcn_wmma_f32_16x16x32_f16(false, af0, false, bf1,
                                                   (short)0, acc01, false, false);
    acc10 = __builtin_amdgcn_wmma_f32_16x16x32_f16(false, af1, false, bf0,
                                                   (short)0, acc10, false, false);
    acc11 = __builtin_amdgcn_wmma_f32_16x16x32_f16(false, af1, false, bf1,
                                                   (short)0, acc11, false, false);
    __syncthreads();
  }
  (void)stepA; (void)stepB;

  const float* nAb = nAq ? (nAq + h * naSh + o * naSo) : nullptr;
  const float* nBb = nBq ? (nBq + h * nbSh + o * nbSo) : nullptr;
  const float sf2h = (mode == 1) ? sf2v[h] : 0.f;
#pragma unroll
  for (int sub = 0; sub < 4; ++sub) {
    const v8f* accp = (sub == 0) ? &acc00 : (sub == 1) ? &acc01
                     : (sub == 2) ? &acc10 : &acc11;
    int subm = (sub >> 1) * 16, subn = (sub & 1) * 16;
#pragma unroll
    for (int r = 0; r < 8; ++r) {
      int gm = m0 + wm + subm + 8 * hi + r;
      int gn = n0 + wn + subn + l16;
      if (gm < Mrows && gn < Ncols) {
        float s = (*accp)[r], val;
        if (mode == 1) {
          float d2 = nAb[gm] + nBb[gn] - 2.f * s;
          val = sf2h * expf(-0.5f * fmaxf(d2, 0.f));
          if (gm == gn) val += jitter;
        } else {
          val = s;
          if (gm == gn) val += jitter;
          if (accumulate) val += Cb[(long)gm * ldC + gn];
        }
        Cb[(long)gm * ldC + gn] = val;
      }
    }
  }
}

// ---------------------------------------------------------------------------
// In-place Cholesky (lower) of an N x N row-major matrix per batch; zeros the
// strict upper triangle afterwards (L_joint is later used as a solve RHS).
// ---------------------------------------------------------------------------
__global__ __launch_bounds__(256)
void cholesky_kernel(float* __restrict__ A, int N, long batchStride) {
  float* Ab = A + (long)blockIdx.x * batchStride;
  const int tid = threadIdx.x, nt = blockDim.x;
  for (int k = 0; k < N; ++k) {
    if (tid == 0) Ab[(long)k * N + k] = sqrtf(Ab[(long)k * N + k]);
    __syncthreads();
    float dinv = 1.f / Ab[(long)k * N + k];
    for (int i = k + 1 + tid; i < N; i += nt) Ab[(long)i * N + k] *= dinv;
    __syncthreads();
    for (int i = k + 1 + tid; i < N; i += nt) {
      float lik = Ab[(long)i * N + k];
      for (int j = k + 1; j <= i; ++j)
        Ab[(long)i * N + j] -= lik * Ab[(long)j * N + k];
    }
    __syncthreads();
  }
  for (int idx = tid; idx < N * N; idx += nt) {
    int i = idx / N, j = idx % N;
    if (j > i) Ab[idx] = 0.f;
  }
}

// ---------------------------------------------------------------------------
// In-place forward substitution  X = L^{-1} X,  L lower NxN, X is N x K
// (row-major, ld = K).  One thread per column -> independent, no syncs.
// Inner recurrence unrolled x4 so loads clause up before one wait.
// ---------------------------------------------------------------------------
__global__ __launch_bounds__(128)
void trsm_kernel(const float* __restrict__ L, float* __restrict__ Xm,
                 int N, int K, long lStride, long xStride) {
  const float* Lb = L + (long)blockIdx.y * lStride;
  float* Xb = Xm + (long)blockIdx.y * xStride;
  int c = blockIdx.x * blockDim.x + threadIdx.x;
  if (c >= K) return;
  for (int i = 0; i < N; ++i) {
    const float* Lrow = Lb + (long)i * N;
    float s = Xb[(long)i * K + c];
    int j = 0;
    float s0 = 0.f, s1 = 0.f, s2 = 0.f, s3 = 0.f;
    const float* px = Xb + c;
    for (; j + 4 <= i; j += 4) {
      float l0 = Lrow[j], l1 = Lrow[j + 1], l2 = Lrow[j + 2], l3 = Lrow[j + 3];
      float x0 = px[0];
      float x1 = px[(long)K];
      float x2 = px[(long)2 * K];
      float x3 = px[(long)3 * K];
      px += (long)4 * K;
      s0 += l0 * x0; s1 += l1 * x1; s2 += l2 * x2; s3 += l3 * x3;
    }
    for (; j < i; ++j) { s0 += Lrow[j] * px[0]; px += (long)K; }
    s -= (s0 + s1) + (s2 + s3);
    Xb[(long)i * K + c] = s / Lrow[i];
  }
}

// ---------------------------------------------------------------------------
// Final fused kernel per (batch, 32-column strip of B):
//   diag1[n] = sum_i X[i,n]^2,   mu[n] = sum_i X[i,n]*u[i]   (VALU, f32)
//   diag2[n] = sum_j (sum_i Ls[i,j]*X[i,n])^2   (WMMA 256x32x256, vec2 never
//   materialized in HBM).   out: mu, then var = sf2 - diag1 + diag2.
// LDS layouts are k-major so fragments come from ds_load_b128.
// ---------------------------------------------------------------------------
__global__ __launch_bounds__(128)
void final_kernel(const float* __restrict__ X,   // LKinvKuf2 [HO][2M][B]
                  const float* __restrict__ Ls,  // LKinvLs2  [HO][2M][2M]
                  const float* __restrict__ U,   // LKinvu    [HO][2M]
                  const float* __restrict__ sf2v,
                  float* __restrict__ out) {
  __shared__ alignas(16) _Float16 lX[32][264];   // [n][i]  (264*2B = 528B rows)
  __shared__ alignas(16) _Float16 lA[32][40];    // [j][i-chunk]
  __shared__ float lu[CM2];
  __shared__ float part_d1[4][32];
  __shared__ float part_mu[4][32];
  __shared__ float accd2[32];

  const int tid = threadIdx.x;
  const int batch = blockIdx.y;
  const int h = batch / CO;
  const int n0 = blockIdx.x * 32;
  const float* Xb  = X  + (long)batch * CM2 * CB;
  const float* Lsb = Ls + (long)batch * CM2 * CM2;
  const float* Ub  = U  + (long)batch * CM2;

  for (int i = tid; i < CM2; i += 128) lu[i] = Ub[i];
  if (tid < 32) accd2[tid] = 0.f;
  __syncthreads();

  // stage X columns (transposed into [n][i]); fold in diag1 & mu partials
  const int n = tid & 31, grp = tid >> 5;       // grp 0..3
  float d1 = 0.f, mu = 0.f;
  for (int ib = 0; ib < 8; ++ib) {
    int i0 = ib * 32 + grp * 8;
    const float* px = Xb + (long)i0 * CB + n0 + n;
    float v[8];
#pragma unroll
    for (int e = 0; e < 8; ++e) { v[e] = *px; px += CB; }
    h8 hv;
#pragma unroll
    for (int e = 0; e < 8; ++e) {
      d1 += v[e] * v[e];
      mu += v[e] * lu[i0 + e];
      hv[e] = (_Float16)v[e];
    }
    *(h8*)&lX[n][i0] = hv;
  }
  part_d1[grp][n] = d1;
  part_mu[grp][n] = mu;
  __syncthreads();

  const int lane = tid & 31, w = tid >> 5;
  const int wm = (w >> 1) * 16, wn = (w & 1) * 16;
  const int l16 = lane & 15, hi = lane >> 4;
  const int sm = tid >> 2, skg = (tid & 3) * 8;  // lA staging: row sm, 8 k's

  for (int jc = 0; jc < CM2 / 32; ++jc) {        // output row chunks of vec2
    v8f acc = {0.f,0.f,0.f,0.f,0.f,0.f,0.f,0.f};
    for (int ic = 0; ic < CM2 / 32; ++ic) {      // reduction chunks over i
      {
        const float* pl = Lsb + (long)(ic * 32 + skg) * CM2 + jc * 32 + sm;
        float fv[8];
#pragma unroll
        for (int e = 0; e < 8; ++e) { fv[e] = *pl; pl += CM2; }
        h8 hv;
#pragma unroll
        for (int e = 0; e < 8; ++e) hv[e] = (_Float16)fv[e];
        *(h8*)&lA[sm][skg] = hv;
      }
      __syncthreads();
      v16h af = frag_from_lds(&lA[wm + l16][0], hi);
      v16h bf = frag_from_lds(&lX[wn + l16][ic * 32], hi);
      acc = __builtin_amdgcn_wmma_f32_16x16x32_f16(false, af, false, bf,
                                                   (short)0, acc, false, false);
      __syncthreads();
    }
    float sq = 0.f;
#pragma unroll
    for (int r = 0; r < 8; ++r) sq += acc[r] * acc[r];
    atomicAdd(&accd2[wn + l16], sq);
  }
  __syncthreads();

  if (tid < 32) {
    float dd1 = part_d1[0][tid] + part_d1[1][tid] + part_d1[2][tid] + part_d1[3][tid];
    float dmu = part_mu[0][tid] + part_mu[1][tid] + part_mu[2][tid] + part_mu[3][tid];
    long oi = (long)batch * CB + n0 + tid;
    out[oi] = dmu;                                             // mu
    out[(long)CH * CO * CB + oi] = sf2v[h] - dd1 + accd2[tid]; // var
  }
}

// ---------------------------------------------------------------------------
extern "C" void kernel_launch(void* const* d_in, const int* in_sizes, int n_in,
                              void* d_out, int out_size, void* d_ws, size_t ws_size,
                              hipStream_t stream) {
  (void)in_sizes; (void)n_in; (void)out_size; (void)ws_size;
  const float* x      = (const float*)d_in[0];   // [B,D]
  const float* z      = (const float*)d_in[1];   // [O,M,D]
  const float* u_mean = (const float*)d_in[2];   // [O,M,1]
  const float* u_tv   = (const float*)d_in[3];   // [O,M(M+1)/2]
  const float* m_old  = (const float*)d_in[4];   // [O,M,1]
  const float* L_old  = (const float*)d_in[5];   // [O,M,M]
  const float* z_old  = (const float*)d_in[6];   // [O,M,D]
  const float* theta  = (const float*)d_in[7];   // [H,D+1]
  float* out = (float*)d_out;
  float* ws  = (float*)d_ws;

  const long HO = (long)CH * CO;

  // ---- prep -----------------------------------------------------------
  prep_theta_kernel<<<1, 64, 0, stream>>>(theta, ws + WS_SF2, ws + WS_LSINV);
  prep_zs_kernel<<<(CH * CO * CM2 + 255) / 256, 256, 0, stream>>>(
      z, z_old, ws + WS_LSINV, ws + WS_ZS, ws + WS_NZ);
  prep_x_kernel<<<(CH * CB + 255) / 256, 256, 0, stream>>>(
      x, ws + WS_LSINV, ws + WS_XSC, ws + WS_NX);
  prep_utril_kernel<<<(CO * CM + 255) / 256, 256, 0, stream>>>(u_tv, ws + WS_UTRIL);
  bcast_copy_kernel<<<256, 256, 0, stream>>>(m_old, ws + WS_MJ,  (long)CM, (long)CM2, HO * CM);
  bcast_copy_kernel<<<256, 256, 0, stream>>>(m_old, ws + WS_LKM, (long)CM, (long)CM,  HO * CM);
  bcast_copy_kernel<<<1024, 256, 0, stream>>>(L_old, ws + WS_LLS,
                                              (long)CM * CM, (long)CM * CM, HO * CM * CM);

  const dim3 g22(2, 2, CH * CO);                 // 128x128 outputs, 64x64 tiles
  const long zsH = (long)CO * CM2 * CDP, zsO = (long)CM2 * CDP;
  const long nzH = (long)CO * CM2,     nzO = (long)CM2;
  const long mmH = (long)CO * CM * CM, mmO = (long)CM * CM;
  const long cvH = (long)CO * CM2 * CM2, cvO = (long)CM2 * CM2;

  // ---- phase A: kuu, kuf, Lkuu, solves, AtX, cov_joint ----------------
  // kuu = k(z_old, z_old) + jitter I
  wmma_gemm_kernel<<<g22, 128, 0, stream>>>(ws + WS_ZS, ws + WS_ZS, ws + WS_KUU,
      CM, CM, CDP, zsH, zsO, 1L, (long)CDP, zsH, zsO, 1L, (long)CDP,
      mmH, mmO, (long)CM, 0, JIT, 1,
      ws + WS_NZ, nzH, nzO, ws + WS_NZ, nzH, nzO, ws + WS_SF2);
  // kuf = k(z_old, z)
  wmma_gemm_kernel<<<g22, 128, 0, stream>>>(ws + WS_ZS, ws + WS_ZS + (long)CM * CDP,
      ws + WS_KUF,
      CM, CM, CDP, zsH, zsO, 1L, (long)CDP, zsH, zsO, 1L, (long)CDP,
      mmH, mmO, (long)CM, 0, 0.f, 1,
      ws + WS_NZ, nzH, nzO, ws + WS_NZ + CM, nzH, nzO, ws + WS_SF2);
  cholesky_kernel<<<CH * CO, 256, 0, stream>>>(ws + WS_KUU, CM, mmO);
  trsm_kernel<<<dim3(1, CH * CO), 128, 0, stream>>>(ws + WS_KUU, ws + WS_KUF,
      CM, CM, mmO, mmO);                                   // LKinvKuf
  trsm_kernel<<<dim3(1, CH * CO), 128, 0, stream>>>(ws + WS_KUU, ws + WS_LKM,
      CM, 1, mmO, (long)CM);                               // LKinvm
  trsm_kernel<<<dim3(1, CH * CO), 128, 0, stream>>>(ws + WS_KUU, ws + WS_LLS,
      CM, CM, mmO, mmO);                                   // LKinvLS
  // Am = LKinvKuf^T @ LKinvm  (Ncols=1, sBn=0 -> column reads alias col 0)
  wmma_gemm_kernel<<<dim3(2, 1, CH * CO), 128, 0, stream>>>(ws + WS_KUF, ws + WS_LKM,
      ws + WS_AM,
      CM, 1, CM, mmH, mmO, (long)CM, 1L, (long)CO * CM, (long)CM, 1L, 0L,
      (long)CO * CM, (long)CM, 1L, 0, 0.f, 0,
      nullptr, 0, 0, nullptr, 0, 0, nullptr);
  add_mnew_kernel<<<(CH * CO * CM + 255) / 256, 256, 0, stream>>>(
      ws + WS_AM, u_mean, ws + WS_MJ);
  // AtX = LKinvLS^T @ LKinvKuf
  wmma_gemm_kernel<<<g22, 128, 0, stream>>>(ws + WS_LLS, ws + WS_KUF, ws + WS_ATX,
      CM, CM, CM, mmH, mmO, (long)CM, 1L, mmH, mmO, (long)CM, 1L,
      mmH, mmO, (long)CM, 0, 0.f, 0,
      nullptr, 0, 0, nullptr, 0, 0, nullptr);
  // cov_joint quadrants (ld = 2M)
  // C11 = L_old L_old^T + jitter I
  wmma_gemm_kernel<<<g22, 128, 0, stream>>>(L_old, L_old, ws + WS_COV,
      CM, CM, CM, 0L, mmO, 1L, (long)CM, 0L, mmO, 1L, (long)CM,
      cvH, cvO, (long)CM2, 0, JIT, 0,
      nullptr, 0, 0, nullptr, 0, 0, nullptr);
  // C12 = L_old @ AtX
  wmma_gemm_kernel<<<g22, 128, 0, stream>>>(L_old, ws + WS_ATX, ws + WS_COV + CM,
      CM, CM, CM, 0L, mmO, 1L, (long)CM, mmH, mmO, (long)CM, 1L,
      cvH, cvO, (long)CM2, 0, 0.f, 0,
      nullptr, 0, 0, nullptr, 0, 0, nullptr);
  // C21 = AtX^T @ L_old^T
  wmma_gemm_kernel<<<g22, 128, 0, stream>>>(ws + WS_ATX, L_old,
      ws + WS_COV + (long)CM * CM2,
      CM, CM, CM, mmH, mmO, (long)CM, 1L, 0L, mmO, 1L, (long)CM,
      cvH, cvO, (long)CM2, 0, 0.f, 0,
      nullptr, 0, 0, nullptr, 0, 0, nullptr);
  // C22 = u_tril u_tril^T + jitter I, then += AtX^T AtX
  wmma_gemm_kernel<<<g22, 128, 0, stream>>>(ws + WS_UTRIL, ws + WS_UTRIL,
      ws + WS_COV + (long)CM * CM2 + CM,
      CM, CM, CM, 0L, mmO, 1L, (long)CM, 0L, mmO, 1L, (long)CM,
      cvH, cvO, (long)CM2, 0, JIT, 0,
      nullptr, 0, 0, nullptr, 0, 0, nullptr);
  wmma_gemm_kernel<<<g22, 128, 0, stream>>>(ws + WS_ATX, ws + WS_ATX,
      ws + WS_COV + (long)CM * CM2 + CM,
      CM, CM, CM, mmH, mmO, (long)CM, 1L, mmH, mmO, (long)CM, 1L,
      cvH, cvO, (long)CM2, 1, 0.f, 0,
      nullptr, 0, 0, nullptr, 0, 0, nullptr);
  cholesky_kernel<<<CH * CO, 256, 0, stream>>>(ws + WS_COV, CM2, cvO);   // L_joint

  // ---- phase B: kuu2, Lk2, solves, kuf2, final ------------------------
  wmma_gemm_kernel<<<dim3(4, 4, CH * CO), 128, 0, stream>>>(ws + WS_ZS, ws + WS_ZS,
      ws + WS_KUU2,
      CM2, CM2, CDP, zsH, zsO, 1L, (long)CDP, zsH, zsO, 1L, (long)CDP,
      cvH, cvO, (long)CM2, 0, JIT, 1,
      ws + WS_NZ, nzH, nzO, ws + WS_NZ, nzH, nzO, ws + WS_SF2);
  cholesky_kernel<<<CH * CO, 256, 0, stream>>>(ws + WS_KUU2, CM2, cvO);  // Lk2
  trsm_kernel<<<dim3(1, CH * CO), 128, 0, stream>>>(ws + WS_KUU2, ws + WS_MJ,
      CM2, 1, cvO, (long)CM2);                              // LKinvu
  // kuf2 = k(z_joint, x)   [HO][2M][B]
  wmma_gemm_kernel<<<dim3(4, 16, CH * CO), 128, 0, stream>>>(ws + WS_ZS, ws + WS_XSC,
      ws + WS_KUF2,
      CM2, CB, CDP, zsH, zsO, 1L, (long)CDP, (long)CB * CDP, 0L, 1L, (long)CDP,
      (long)CO * CM2 * CB, (long)CM2 * CB, (long)CB, 0, 0.f, 1,
      ws + WS_NZ, nzH, nzO, ws + WS_NX, (long)CB, 0L, ws + WS_SF2);
  trsm_kernel<<<dim3(8, CH * CO), 128, 0, stream>>>(ws + WS_KUU2, ws + WS_KUF2,
      CM2, CB, cvO, (long)CM2 * CB);                        // LKinvKuf2
  trsm_kernel<<<dim3(2, CH * CO), 128, 0, stream>>>(ws + WS_KUU2, ws + WS_COV,
      CM2, CM2, cvO, cvO);                                  // LKinvLs2
  final_kernel<<<dim3(CB / 32, CH * CO), 128, 0, stream>>>(
      ws + WS_KUF2, ws + WS_COV, ws + WS_MJ, ws + WS_SF2, out);
}
